// expand_st_adapter_62981400429138
// MI455X (gfx1250) — compile-verified, matrix operand order checked by reading, not run on hardware
//
#include <hip/hip_runtime.h>
#include <hip/hip_bf16.h>
#include <math.h>

// ---------------- CDNA5 WMMA types ----------------
typedef __attribute__((ext_vector_type(16))) __bf16 v16bf;
typedef __attribute__((ext_vector_type(8)))  float  v8f;
typedef __attribute__((ext_vector_type(4)))  float  f4;

// ---------------- problem constants ----------------
#define CDIM   768
#define LSEQ   197
#define LPAD   224          // 7 * 32 : padded attention extent (K-safe)
#define NSEQ   128          // B * ngroups * 4 = 16*2*4
#define NHEADS 12
#define HDIM   64
#define FFDIM  3072
#define NLAYER 12
#define NCLS   400
#define NCLSP  448          // 7 * 64 : padded head rows
#define NROWS  (NSEQ * LSEQ)     // 25216 = 197 * 128 (exact M tiles)
#define PEROWS (128 * 196)       // 25088 = 196 * 128 (exact M tiles)
#define QKVPAD 128               // extra qkv rows so M/N over-reads stay in-bounds

struct GemmP {
  const float* A; const float* B; float* C;
  const float* bias; const float* resid;
  int M, N, K;
  int lda, ldb, ldc;
  long long aO, aI, bO, bI, cO, cI;   // batch offsets (elements)
  int nbInner;
  float alpha;
  int act;       // 0 none, 1: x*sigmoid(1.702x)
};

// =====================================================================
// FAST batched GEMM: C = act(alpha * A @ op(B) + bias) (+ residual)
// Requirements: K % 32 == 0; M/N over-reads land in padded memory
// (over-read rows/cols are discarded by the guarded epilogue).
// 256 threads = 8 waves (wave32); block tile 128x64, wave tile 32x32.
// Double-buffered LDS software pipeline: next tile's global loads are
// issued before the current tile's WMMAs -> HBM latency hides under math.
// =====================================================================
template <bool TRANSB>
__global__ __launch_bounds__(256) void gemm_fast(GemmP p) {
  __shared__ __bf16 As[2][128][36];
  __shared__ __bf16 Bs[2][64][36];

  const int tid  = threadIdx.x;
  const int wave = tid >> 5, lane = tid & 31;
  const int half = lane >> 4, lm = lane & 15;
  const int waveM = (wave >> 1) * 32, waveN = (wave & 1) * 32;
  const int blockM = blockIdx.y * 128, blockN = blockIdx.x * 64;

  const int bz = blockIdx.z;
  const int outer = bz / p.nbInner, inner = bz % p.nbInner;
  const long long offA = p.aO * outer + p.aI * inner;
  const long long offB = p.bO * outer + p.bI * inner;
  const long long offC = p.cO * outer + p.cI * inner;

  const int ra = tid >> 3;          // 0..31
  const int c4 = (tid & 7) * 4;     // fixed 16B column base per thread
  const int kk2 = tid >> 4;         // transB==false B indexing
  const int n4a = (tid & 15) * 4;
  const int kk2b = (tid >> 4) + 16;

  v8f acc[2][2] = {};
  f4 av[4], bv[2];

  const int steps = p.K >> 5;       // K % 32 == 0 guaranteed

  // ---- preload tile 0 into registers ----
#pragma unroll
  for (int i = 0; i < 4; ++i)
    av[i] = *(const f4*)&p.A[offA + (long long)(blockM + ra + 32 * i) * p.lda + c4];
  if (TRANSB) {
#pragma unroll
    for (int i = 0; i < 2; ++i)
      bv[i] = *(const f4*)&p.B[offB + (long long)(blockN + ra + 32 * i) * p.ldb + c4];
  } else {
    bv[0] = *(const f4*)&p.B[offB + (long long)kk2  * p.ldb + blockN + n4a];
    bv[1] = *(const f4*)&p.B[offB + (long long)kk2b * p.ldb + blockN + n4a];
  }

  for (int s = 0; s < steps; ++s) {
    const int buf = s & 1;
    // ---- commit staged registers to LDS (bf16) ----
#pragma unroll
    for (int i = 0; i < 4; ++i) {
      union { __bf16 h[4]; unsigned long long u; } pk;
      pk.h[0] = (__bf16)av[i][0]; pk.h[1] = (__bf16)av[i][1];
      pk.h[2] = (__bf16)av[i][2]; pk.h[3] = (__bf16)av[i][3];
      *(unsigned long long*)&As[buf][ra + 32 * i][c4] = pk.u;
    }
    if (TRANSB) {
#pragma unroll
      for (int i = 0; i < 2; ++i) {
        union { __bf16 h[4]; unsigned long long u; } pk;
        pk.h[0] = (__bf16)bv[i][0]; pk.h[1] = (__bf16)bv[i][1];
        pk.h[2] = (__bf16)bv[i][2]; pk.h[3] = (__bf16)bv[i][3];
        *(unsigned long long*)&Bs[buf][ra + 32 * i][c4] = pk.u;
      }
    } else {
#pragma unroll
      for (int j = 0; j < 4; ++j) Bs[buf][n4a + j][kk2]  = (__bf16)bv[0][j];
#pragma unroll
      for (int j = 0; j < 4; ++j) Bs[buf][n4a + j][kk2b] = (__bf16)bv[1][j];
    }
    __syncthreads();

    // ---- issue next tile's global loads (overlap with WMMA below) ----
    if (s + 1 < steps) {
      const int k0 = (s + 1) << 5;
#pragma unroll
      for (int i = 0; i < 4; ++i)
        av[i] = *(const f4*)&p.A[offA + (long long)(blockM + ra + 32 * i) * p.lda + k0 + c4];
      if (TRANSB) {
#pragma unroll
        for (int i = 0; i < 2; ++i)
          bv[i] = *(const f4*)&p.B[offB + (long long)(blockN + ra + 32 * i) * p.ldb + k0 + c4];
      } else {
        bv[0] = *(const f4*)&p.B[offB + (long long)(k0 + kk2)  * p.ldb + blockN + n4a];
        bv[1] = *(const f4*)&p.B[offB + (long long)(k0 + kk2b) * p.ldb + blockN + n4a];
      }
      if (s + 2 < steps)  // speculative prefetch two tiles ahead
        __builtin_prefetch(&p.A[offA + (long long)(blockM + ra) * p.lda + ((s + 2) << 5) + c4], 0, 0);
    }

    // ---- fragments per ISA 7.12.2 (16-bit A 16x32 layout) + WMMA ----
    v16bf a0, a1, b0, b1;
#pragma unroll
    for (int i = 0; i < 8; ++i) {
      int kk = ((i & 3) << 1) + ((i >> 2) << 4) + (half << 3);
      a0[2 * i]     = As[buf][waveM + lm][kk];
      a0[2 * i + 1] = As[buf][waveM + lm][kk + 1];
      a1[2 * i]     = As[buf][waveM + 16 + lm][kk];
      a1[2 * i + 1] = As[buf][waveM + 16 + lm][kk + 1];
      b0[2 * i]     = Bs[buf][waveN + lm][kk];
      b0[2 * i + 1] = Bs[buf][waveN + lm][kk + 1];
      b1[2 * i]     = Bs[buf][waveN + 16 + lm][kk];
      b1[2 * i + 1] = Bs[buf][waveN + 16 + lm][kk + 1];
    }
    acc[0][0] = __builtin_amdgcn_wmma_f32_16x16x32_bf16(false, a0, false, b0, (short)0, acc[0][0], false, false);
    acc[0][1] = __builtin_amdgcn_wmma_f32_16x16x32_bf16(false, a0, false, b1, (short)0, acc[0][1], false, false);
    acc[1][0] = __builtin_amdgcn_wmma_f32_16x16x32_bf16(false, a1, false, b0, (short)0, acc[1][0], false, false);
    acc[1][1] = __builtin_amdgcn_wmma_f32_16x16x32_bf16(false, a1, false, b1, (short)0, acc[1][1], false, false);
  }

  // ---- epilogue: D layout lanes 0-15: M=r, lanes 16-31: M=r+8; N=lane%16
#pragma unroll
  for (int ti = 0; ti < 2; ++ti)
#pragma unroll
    for (int tj = 0; tj < 2; ++tj)
#pragma unroll
      for (int r = 0; r < 8; ++r) {
        int row = blockM + waveM + ti * 16 + half * 8 + r;
        int col = blockN + waveN + tj * 16 + lm;
        if (row < p.M && col < p.N) {
          float v = p.alpha * acc[ti][tj][r];
          if (p.bias) v += p.bias[col];
          if (p.act == 1) v = v * (1.0f / (1.0f + __expf(-1.702f * v)));
          long long ci = offC + (long long)row * p.ldc + col;
          if (p.resid) v += p.resid[ci];
          p.C[ci] = v;
        }
      }
}

// =====================================================================
// LayerNorm over 768 features (one block per row)
// =====================================================================
__global__ __launch_bounds__(256) void ln768_kernel(const float* __restrict__ in,
                                                    float* __restrict__ out,
                                                    const float* __restrict__ w,
                                                    const float* __restrict__ b,
                                                    int rows) {
  int row = blockIdx.x;
  if (row >= rows) return;
  const float* x = in + (long long)row * CDIM;
  int t = threadIdx.x;
  float l0 = x[t], l1 = x[t + 256], l2 = x[t + 512];
  __shared__ float red[256];
  red[t] = l0 + l1 + l2;
  __syncthreads();
  for (int st = 128; st > 0; st >>= 1) { if (t < st) red[t] += red[t + st]; __syncthreads(); }
  float mean = red[0] * (1.0f / CDIM);
  __syncthreads();
  float d0 = l0 - mean, d1 = l1 - mean, d2 = l2 - mean;
  red[t] = d0 * d0 + d1 * d1 + d2 * d2;
  __syncthreads();
  for (int st = 128; st > 0; st >>= 1) { if (t < st) red[t] += red[t + st]; __syncthreads(); }
  float rstd = rsqrtf(red[0] * (1.0f / CDIM) + 1e-5f);
  float* o = out + (long long)row * CDIM;
  o[t]       = d0 * rstd * w[t]       + b[t];
  o[t + 256] = d1 * rstd * w[t + 256] + b[t + 256];
  o[t + 512] = d2 * rstd * w[t + 512] + b[t + 512];
}

// =====================================================================
// Row softmax over 197 valid cols (ld = LPAD); zero-fills cols 197..LPAD-1
// so the AV GEMM can run an unguarded K loop over LPAD.
// =====================================================================
__global__ __launch_bounds__(256) void softmax197_kernel(float* __restrict__ scores) {
  int row = blockIdx.x;          // 0..196
  int batch = blockIdx.y;        // 0..1535
  float* ptr = scores + (long long)batch * (LPAD * LPAD) + (long long)row * LPAD;
  int t = threadIdx.x;
  float v = (t < LSEQ) ? ptr[t] : -3.0e38f;
  __shared__ float red[256];
  red[t] = v;
  __syncthreads();
  for (int st = 128; st > 0; st >>= 1) { if (t < st) red[t] = fmaxf(red[t], red[t + st]); __syncthreads(); }
  float mx = red[0];
  __syncthreads();
  float e = (t < LSEQ) ? __expf(v - mx) : 0.f;
  red[t] = e;
  __syncthreads();
  for (int st = 128; st > 0; st >>= 1) { if (t < st) red[t] += red[t + st]; __syncthreads(); }
  float inv = 1.0f / red[0];
  if (t < LSEQ) ptr[t] = e * inv;
  else if (t < LPAD) ptr[t] = 0.f;
}

// =====================================================================
// im2col for non-overlapping 16x16 patches.
// =====================================================================
__global__ void im2col_kernel(const float* __restrict__ x, float* __restrict__ A) {
  long long total = (long long)PEROWS * CDIM;
  for (long long idx = blockIdx.x * 256ll + threadIdx.x; idx < total;
       idx += (long long)gridDim.x * 256ll) {
    int k = (int)(idx % CDIM);
    long long row = idx / CDIM;
    int bt = (int)(row / 196), s = (int)(row % 196);
    int gh = s / 14, gw = s % 14;
    int c = k >> 8, rem = k & 255, pp = rem >> 4, q = rem & 15;
    int b = bt >> 3, t = bt & 7;
    long long src = ((((long long)b * 3 + c) * 8 + t) * 224 + (gh * 16 + pp)) * 224
                    + (gw * 16 + q);
    A[idx] = x[src];
  }
}

// =====================================================================
// Copy head weights [400,768] into zero-padded [448,768] so the head
// GEMM can use the unguarded fast path.
// =====================================================================
__global__ void pad_head_kernel(const float* __restrict__ w, float* __restrict__ wp) {
  int idx = blockIdx.x * 256 + threadIdx.x;       // NCLSP*CDIM
  if (idx >= NCLSP * CDIM) return;
  int n = idx / CDIM;
  wp[idx] = (n < NCLS) ? w[idx] : 0.f;
}

// =====================================================================
// Build tube index tables on device (tiny, single thread).
// =====================================================================
__global__ void init_tables_kernel(int* __restrict__ pidx, int* __restrict__ foff) {
  if (threadIdx.x != 0 || blockIdx.x != 0) return;
  const int psz[4] = {3, 5, 7, 9};
  const int sts[4] = {45, 48, 87, 90};
  for (int f = 0; f < 4; ++f) {
    int w = 0;
    for (int gi = 0; gi < 4; ++gi) {
      int ps = psz[gi];
      int sp = sts[f] - 15 * ((ps - 3) / 2);
      int g = (ps + 1) / 2;
      int tmp[89];
      int n = 0;
      tmp[n++] = sp;             tmp[n++] = sp + g;            tmp[n++] = sp + 2 * g;
      tmp[n++] = sp + 14 * g;    tmp[n++] = sp + 14 * g + 2 * g;
      tmp[n++] = sp + 28 * g;    tmp[n++] = sp + 28 * g + g;   tmp[n++] = sp + 28 * g + 2 * g;
      for (int j = 0; j < ps; ++j)
        for (int i = 0; i < ps; ++i) tmp[n++] = 14 * i + sp + j + 1;
      for (int a = 1; a < n; ++a) {
        int key = tmp[a]; int bq = a - 1;
        while (bq >= 0 && tmp[bq] > key) { tmp[bq + 1] = tmp[bq]; --bq; }
        tmp[bq + 1] = key;
      }
      for (int a = 0; a < n; ++a) {
        pidx[f * 196 + w] = tmp[a];
        if (f == 0) foff[w] = gi;
        ++w;
      }
    }
  }
}

// =====================================================================
// Gather tubes + cls + positional embeddings -> h [25216, 768]
// =====================================================================
__global__ void assemble_kernel(const float* __restrict__ tokraw,
                                const float* __restrict__ cls,
                                const float* __restrict__ spos,
                                const float* __restrict__ tpos,
                                const int* __restrict__ pidx,
                                const int* __restrict__ foff,
                                float* __restrict__ h) {
  long long total = (long long)NROWS * CDIM;
  for (long long idx = blockIdx.x * 256ll + threadIdx.x; idx < total;
       idx += (long long)gridDim.x * 256ll) {
    int c = (int)(idx % CDIM);
    long long row = idx / CDIM;
    int tok = (int)(row % LSEQ);
    int seq = (int)(row / LSEQ);
    float v;
    if (tok == 0) {
      v = cls[c];
    } else {
      int j = tok - 1;
      int f = seq & 3, g = (seq >> 2) & 1, b = seq >> 3;
      int t = 4 * g + foff[j];
      int sp = pidx[f * 196 + j];
      v = tokraw[((long long)((b * 8 + t) * 196 + sp)) * CDIM + c]
        + spos[(long long)sp * CDIM + c]
        + tpos[(long long)t * (196 * CDIM) + (long long)sp * CDIM + c];
    }
    h[idx] = v;
  }
}

// =====================================================================
// cls pooling over 8 frames per batch
// =====================================================================
__global__ void pool_cls_kernel(const float* __restrict__ h, float* __restrict__ pooled) {
  int idx = blockIdx.x * 256 + threadIdx.x;   // 16*768
  if (idx >= 16 * CDIM) return;
  int b = idx / CDIM, c = idx % CDIM;
  float s = 0.f;
  for (int t = 0; t < 8; ++t)
    s += h[((long long)(b * 8 + t) * LSEQ) * CDIM + c];
  pooled[idx] = s * 0.125f;
}

// =====================================================================
// host-side GEMM launch helper
// =====================================================================
static inline void launch_gemm(hipStream_t st, const float* A, const float* B, float* C,
                               const float* bias, const float* resid,
                               int M, int N, int K, int lda, int ldb, int ldc,
                               int transB, float alpha, int act,
                               int nb = 1, int nbInner = 1,
                               long long aO = 0, long long aI = 0,
                               long long bO = 0, long long bI = 0,
                               long long cO = 0, long long cI = 0) {
  GemmP p;
  p.A = A; p.B = B; p.C = C; p.bias = bias; p.resid = resid;
  p.M = M; p.N = N; p.K = K; p.lda = lda; p.ldb = ldb; p.ldc = ldc;
  p.aO = aO; p.aI = aI; p.bO = bO; p.bI = bI; p.cO = cO; p.cI = cI;
  p.nbInner = nbInner; p.alpha = alpha; p.act = act;
  dim3 grid((N + 63) / 64, (M + 127) / 128, nb);
  if (transB) gemm_fast<true><<<grid, 256, 0, st>>>(p);
  else        gemm_fast<false><<<grid, 256, 0, st>>>(p);
}

// =====================================================================
extern "C" void kernel_launch(void* const* d_in, const int* in_sizes, int n_in,
                              void* d_out, int out_size, void* d_ws, size_t ws_size,
                              hipStream_t stream) {
  const float* x          = (const float*)d_in[0];
  const float* conv1_w    = (const float*)d_in[1];
  const float* class_emb  = (const float*)d_in[2];
  const float* spatial    = (const float*)d_in[3];
  const float* temporal   = (const float*)d_in[4];
  const float* ln_pre_w   = (const float*)d_in[5];
  const float* ln_pre_b   = (const float*)d_in[6];
  const float* in_proj_w  = (const float*)d_in[7];
  const float* in_proj_b  = (const float*)d_in[8];
  const float* out_proj_w = (const float*)d_in[9];
  const float* out_proj_b = (const float*)d_in[10];
  const float* ln1_w      = (const float*)d_in[11];
  const float* ln1_b      = (const float*)d_in[12];
  const float* fc_w       = (const float*)d_in[13];
  const float* fc_b       = (const float*)d_in[14];
  const float* proj_w     = (const float*)d_in[15];
  const float* proj_b     = (const float*)d_in[16];
  const float* ln2_w      = (const float*)d_in[17];
  const float* ln2_b      = (const float*)d_in[18];
  const float* ln_post_w  = (const float*)d_in[19];
  const float* ln_post_b  = (const float*)d_in[20];
  const float* head_w     = (const float*)d_in[21];
  const float* head_b     = (const float*)d_in[22];
  float* out = (float*)d_out;

  // ---------------- workspace layout ----------------
  char* wsb = (char*)d_ws;
  int*   pidx    = (int*)wsb;
  int*   foff    = pidx + 4 * 196;
  float* h       = (float*)(wsb + 4096);              // 25216*768
  float* xs      = h + (long long)NROWS * CDIM;       // 25216*768
  float* attnout = xs + (long long)NROWS * CDIM;      // 25216*768 (tokraw pre-loop)
  float* qkv     = attnout + (long long)NROWS * CDIM; // (25216+128)*2304 padded
  float* big     = qkv + (long long)(NROWS + QKVPAD) * 3 * CDIM;
  float* pooled  = big + (long long)NROWS * FFDIM;    // 16*768
  float* pooledn = pooled + 16 * CDIM;                // 128*768 (padded rows)
  float* headwp  = pooledn + 128 * CDIM;              // 448*768 zero-padded head_w

  float* Ape    = qkv;
  float* tokraw = attnout;
  float* scores = big;      // 1536 * 224*224
  float* ymlp   = big;      // 25216 * 3072 (sizes region)

  const long long scoreB = (long long)LPAD * LPAD;

  // ---------------- pre-loop ----------------
  init_tables_kernel<<<1, 1, 0, stream>>>(pidx, foff);
  pad_head_kernel<<<(NCLSP * CDIM + 255) / 256, 256, 0, stream>>>(head_w, headwp);
  im2col_kernel<<<75264, 256, 0, stream>>>(x, Ape);
  launch_gemm(stream, Ape, conv1_w, tokraw, nullptr, nullptr,
              PEROWS, CDIM, CDIM, CDIM, CDIM, CDIM, 1, 1.0f, 0);
  assemble_kernel<<<75648, 256, 0, stream>>>(tokraw, class_emb, spatial, temporal,
                                             pidx, foff, h);
  ln768_kernel<<<NROWS, 256, 0, stream>>>(h, h, ln_pre_w, ln_pre_b, NROWS);

  // ---------------- transformer layers ----------------
  for (int i = 0; i < NLAYER; ++i) {
    const float* wqkv = in_proj_w  + (long long)i * 3 * CDIM * CDIM;
    const float* bqkv = in_proj_b  + (long long)i * 3 * CDIM;
    const float* wo   = out_proj_w + (long long)i * CDIM * CDIM;
    const float* bo   = out_proj_b + (long long)i * CDIM;
    const float* wfc  = fc_w       + (long long)i * FFDIM * CDIM;
    const float* bfc  = fc_b       + (long long)i * FFDIM;
    const float* wpr  = proj_w     + (long long)i * CDIM * FFDIM;
    const float* bpr  = proj_b     + (long long)i * CDIM;

    ln768_kernel<<<NROWS, 256, 0, stream>>>(h, xs, ln1_w + i * CDIM, ln1_b + i * CDIM, NROWS);
    launch_gemm(stream, xs, wqkv, qkv, bqkv, nullptr,
                NROWS, 3 * CDIM, CDIM, CDIM, CDIM, 3 * CDIM, 1, 1.0f, 0);
    launch_gemm(stream, qkv, qkv + CDIM, scores, nullptr, nullptr,
                LSEQ, LSEQ, HDIM, 3 * CDIM, 3 * CDIM, LPAD, 1,
                0.125f, 0, NSEQ * NHEADS, NHEADS,
                (long long)LSEQ * 3 * CDIM, HDIM,
                (long long)LSEQ * 3 * CDIM, HDIM,
                (long long)NHEADS * scoreB, scoreB);
    softmax197_kernel<<<dim3(LSEQ, NSEQ * NHEADS), 256, 0, stream>>>(scores);
    launch_gemm(stream, scores, qkv + 2 * CDIM, attnout, nullptr, nullptr,
                LSEQ, HDIM, LPAD, LPAD, 3 * CDIM, CDIM, 0,
                1.0f, 0, NSEQ * NHEADS, NHEADS,
                (long long)NHEADS * scoreB, scoreB,
                (long long)LSEQ * 3 * CDIM, HDIM,
                (long long)LSEQ * CDIM, HDIM);
    launch_gemm(stream, attnout, wo, h, bo, h,
                NROWS, CDIM, CDIM, CDIM, CDIM, CDIM, 1, 1.0f, 0);
    ln768_kernel<<<NROWS, 256, 0, stream>>>(h, xs, ln2_w + i * CDIM, ln2_b + i * CDIM, NROWS);
    launch_gemm(stream, xs, wfc, ymlp, bfc, nullptr,
                NROWS, FFDIM, CDIM, CDIM, CDIM, FFDIM, 1, 1.0f, 1);
    launch_gemm(stream, ymlp, wpr, h, bpr, h,
                NROWS, CDIM, FFDIM, FFDIM, FFDIM, CDIM, 1, 1.0f, 0);
  }

  // ---------------- head ----------------
  pool_cls_kernel<<<48, 256, 0, stream>>>(h, pooled);
  ln768_kernel<<<16, 256, 0, stream>>>(pooled, pooledn, ln_post_w, ln_post_b, 16);
  // fast path: M over-reads hit pooledn pad rows, N over-reads hit headwp pad
  launch_gemm(stream, pooledn, headwp, out, head_b, nullptr,
              16, NCLS, CDIM, CDIM, CDIM, NCLS, 1, 1.0f, 0);
}
